// FeaturePyramidNetworks_47863115546759
// MI455X (gfx1250) — matrix operand order repaired and checked by reading, not compiled, not run on hardware
//
#include <hip/hip_runtime.h>

// ---------------------------------------------------------------------------
// CDNA5 (gfx1250, wave32) deformable-conv 3D U-Net.
// All channel contractions via v_wmma_f32_16x16x32_bf16 (f32 accumulate).
// Workspace: ~57 MiB (7 activation slots + argmax indices + packed weights).
// ---------------------------------------------------------------------------

typedef __attribute__((ext_vector_type(16))) __bf16 bf16x16;
typedef __attribute__((ext_vector_type(8)))  float  f32x8;

static __device__ __forceinline__ unsigned short f2bfu(float f) {
  // round-to-nearest-even fp32 -> bf16 bits
  unsigned u = __builtin_bit_cast(unsigned, f);
  return (unsigned short)((u + 0x7FFFu + ((u >> 16) & 1u)) >> 16);
}
static __device__ __forceinline__ __bf16 f2bf(float f) {
  unsigned short s = f2bfu(f);
  return __builtin_bit_cast(__bf16, s);
}
static __device__ __forceinline__ unsigned pack2bf(float v0, float v1) {
  return (unsigned)f2bfu(v0) | ((unsigned)f2bfu(v1) << 16);
}

// ---------------------------------------------------------------------------
// Weight packing: [O(<=64 pad), C=64, 3,3,3] fp32 -> per-tap A fragments.
// A-fragment (16x32 bf16) lane/element mapping used here (contiguous k, B uses the same):
//   lane l: M = 16*mt + (l&15), half = l>>4
//   element e: K = 16*half + e
// Stored [tap][mtile][kchunk][lane][16] -> 32B contiguous per lane.
// ---------------------------------------------------------------------------
__global__ __launch_bounds__(512)
void pack_w_k(const float* __restrict__ w, __bf16* __restrict__ out, int O) {
  int b = blockIdx.x;                   // 27 taps * 4 mtiles * 2 kchunks = 216
  int tap = b >> 3, mt = (b >> 1) & 3, kc = b & 1;
  int t = threadIdx.x;                  // 512 = 32 lanes * 16 elems
  int lane = t >> 4, e = t & 15;
  int half = lane >> 4;
  int m = 16 * mt + (lane & 15);
  int K = 16 * half + e;
  int c = 32 * kc + K;
  float v = (m < O) ? w[(m * 64 + c) * 27 + tap] : 0.f;
  out[(size_t)(((tap * 4 + mt) * 2 + kc) * 32 + lane) * 16 + e] = f2bf(v);
}

// ---------------------------------------------------------------------------
// Per-tap WMMA step. B tile in LDS is fragment-ordered:
//   ldsB[kc*512 + lane*16 + e] <-> (c = 32*kc + 16*(lane>>4) + e, n = lane&15)
// i.e. for element (c,n): j = (c>>4)*256 + n*16 + (c&15).
// ---------------------------------------------------------------------------
static __device__ __forceinline__ void wmma_tap(const __bf16* lds,
                                                const __bf16* __restrict__ wpack,
                                                int tap, int wave, int lane,
                                                f32x8& acc) {
  bf16x16 b0 = *(const bf16x16*)(lds + lane * 16);
  bf16x16 b1 = *(const bf16x16*)(lds + 512 + lane * 16);
  const __bf16* abase = wpack + (size_t)((tap * 4 + wave) * 2 * 32 + lane) * 16;
  bf16x16 a0 = *(const bf16x16*)abase;
  bf16x16 a1 = *(const bf16x16*)(abase + 512);
  acc = __builtin_amdgcn_wmma_f32_16x16x32_bf16(false, a0, false, b0,
                                                (short)0, acc, false, false);
  acc = __builtin_amdgcn_wmma_f32_16x16x32_bf16(false, a1, false, b1,
                                                (short)0, acc, false, false);
}

static __device__ __forceinline__ void epilogue(f32x8 acc, float* __restrict__ dst,
                                                const float* __restrict__ bias,
                                                int bias_len,
                                                const float* __restrict__ resid,
                                                int do_lrelu, int wave, int half,
                                                int nn, int p0, int THW) {
#pragma unroll
  for (int r = 0; r < 8; ++r) {
    int M = 16 * wave + r + 8 * half;      // C/D layout per ISA 7.12.2
    float v = acc[r];
    if (bias) v += (M < bias_len) ? bias[M] : 0.f;
    if (do_lrelu) v = (v >= 0.f) ? v : 0.1f * v;
    int o = M * THW + p0 + nn;
    if (resid) v += resid[o];
    dst[o] = v;
  }
}

// ---------------------------------------------------------------------------
// Plain 3x3x3 conv, 64 -> 64(pad) channels, implicit GEMM over K = 64*27.
// Block = 4 waves = 64x16 output tile; grid.x = T*H*W/16 (always exact).
// Each thread stages column n = tid&15 for channels c = 8*(tid>>4)+i.
// ---------------------------------------------------------------------------
__global__ __launch_bounds__(128)
void conv64_wmma_k(const float* __restrict__ src, const __bf16* __restrict__ wpack,
                   const float* __restrict__ bias, int bias_len,
                   const float* __restrict__ resid, int do_lrelu,
                   float* __restrict__ dst, int T, int lgH, int lgW) {
  __shared__ __align__(32) __bf16 ldsB[2][1024];
  const int lgHW = lgH + lgW;
  const int HW = 1 << lgHW, THW = T * HW;
  const int H = 1 << lgH, W = 1 << lgW;
  const int tid = threadIdx.x;
  const int wave = tid >> 5, lane = tid & 31;
  const int half = lane >> 4, nn = lane & 15;
  const int p0 = blockIdx.x * 16;
  // fixed column + channel group for this thread (tap-invariant)
  const int n = tid & 15, g = tid >> 4;          // g in 0..7
  const int p = p0 + n;
  const int t = p >> lgHW, r2 = p & (HW - 1);
  const int h = r2 >> lgW, w0c = r2 & (W - 1);
  int cBase[8];
  unsigned ldsJ[4];
#pragma unroll
  for (int i = 0; i < 8; ++i) cBase[i] = (8 * g + i) * THW;
#pragma unroll
  for (int i = 0; i < 8; i += 2) {
    int c = 8 * g + i;
    ldsJ[i >> 1] = (unsigned)((c >> 4) * 256 + n * 16 + (c & 15));
  }
  f32x8 acc = {};
  for (int tap = 0; tap < 27; ++tap) {
    const int dt = tap / 9 - 1, dh = (tap / 3) % 3 - 1, dw = tap % 3 - 1;
    int tt = t + dt, hh = h + dh, ww = w0c + dw;
    bool valid = (unsigned)tt < (unsigned)T && (unsigned)hh < (unsigned)H &&
                 (unsigned)ww < (unsigned)W;
    int colOff = valid ? ((tt << lgHW) + (hh << lgW) + ww) : 0;
    float wc = valid ? 1.f : 0.f;
    __bf16* buf = ldsB[tap & 1];
#pragma unroll
    for (int i = 0; i < 8; i += 2) {
      float v0 = wc * src[cBase[i] + colOff];
      float v1 = wc * src[cBase[i + 1] + colOff];
      *(unsigned*)&buf[ldsJ[i >> 1]] = pack2bf(v0, v1);
    }
    if (tap < 26)  // stream next tap's packed weights (global_prefetch)
      __builtin_prefetch(wpack + (size_t)(((tap + 1) * 4 + wave) * 2 * 32 + lane) * 16, 0, 1);
    __syncthreads();
    wmma_tap(buf, wpack, tap, wave, lane, acc);
  }
  epilogue(acc, dst, bias, bias_len, resid, do_lrelu, wave, half, nn, p0, THW);
}

// ---------------------------------------------------------------------------
// Deformable conv (H,W bilinear, integer T shift). off is 64-row-padded,
// channel (2k+d) at stride THW. Same GEMM skeleton; bilinear params are kept
// entirely in registers (each thread owns one column).
// ---------------------------------------------------------------------------
__global__ __launch_bounds__(128)
void deform_wmma_k(const float* __restrict__ src, const float* __restrict__ off,
                   const __bf16* __restrict__ wpack,
                   const float* __restrict__ resid, int do_lrelu,
                   float* __restrict__ dst, int T, int lgH, int lgW) {
  __shared__ __align__(32) __bf16 ldsB[2][1024];
  const int lgHW = lgH + lgW;
  const int HW = 1 << lgHW, THW = T * HW;
  const int H = 1 << lgH, W = 1 << lgW;
  const int tid = threadIdx.x;
  const int wave = tid >> 5, lane = tid & 31;
  const int half = lane >> 4, nn = lane & 15;
  const int p0 = blockIdx.x * 16;
  const int n = tid & 15, g = tid >> 4;
  const int p = p0 + n;
  const int t = p >> lgHW, r2 = p & (HW - 1);
  const int h = r2 >> lgW, w0c = r2 & (W - 1);
  const float* offp = off + p;
  int cBase[8];
  unsigned ldsJ[4];
#pragma unroll
  for (int i = 0; i < 8; ++i) cBase[i] = (8 * g + i) * THW;
#pragma unroll
  for (int i = 0; i < 8; i += 2) {
    int c = 8 * g + i;
    ldsJ[i >> 1] = (unsigned)((c >> 4) * 256 + n * 16 + (c & 15));
  }
  f32x8 acc = {};
  for (int tap = 0; tap < 27; ++tap) {
    const int dt = tap / 9 - 1, dh = (tap / 3) % 3 - 1, dw = tap % 3 - 1;
    // bilinear sampling parameters for this thread's column (registers only)
    int pt = t + dt;
    bool tv = (unsigned)pt < (unsigned)T;
    int ptc = pt < 0 ? 0 : (pt >= T ? T - 1 : pt);
    float o0 = offp[(2 * tap) * THW];
    float o1 = offp[(2 * tap + 1) * THW];
    float ph = (float)(h + dh) + o0;
    float pw = (float)(w0c + dw) + o1;
    float h0f = floorf(ph), w0f = floorf(pw);
    float fh = ph - h0f, fw = pw - w0f;
    int h0 = (int)h0f, w0 = (int)w0f;
    int tb = ptc << lgHW;
    int ixq[4];
    float wq[4];
    float whv[2] = {1.f - fh, fh};
    float wwv[2] = {1.f - fw, fw};
#pragma unroll
    for (int a = 0; a < 2; ++a)
#pragma unroll
      for (int b = 0; b < 2; ++b) {
        int hh = h0 + a, ww = w0 + b;
        bool valid = tv && (unsigned)hh < (unsigned)H && (unsigned)ww < (unsigned)W;
        int hc = hh < 0 ? 0 : (hh > H - 1 ? H - 1 : hh);
        int wc = ww < 0 ? 0 : (ww > W - 1 ? W - 1 : ww);
        ixq[a * 2 + b] = tb + (hc << lgW) + wc;
        wq[a * 2 + b] = valid ? whv[a] * wwv[b] : 0.f;
      }
    __bf16* buf = ldsB[tap & 1];
#pragma unroll
    for (int i = 0; i < 8; i += 2) {
      const float* s0 = src + cBase[i];
      const float* s1 = src + cBase[i + 1];
      float v0 = wq[0] * s0[ixq[0]] + wq[1] * s0[ixq[1]] +
                 wq[2] * s0[ixq[2]] + wq[3] * s0[ixq[3]];
      float v1 = wq[0] * s1[ixq[0]] + wq[1] * s1[ixq[1]] +
                 wq[2] * s1[ixq[2]] + wq[3] * s1[ixq[3]];
      *(unsigned*)&buf[ldsJ[i >> 1]] = pack2bf(v0, v1);
    }
    if (tap < 26)
      __builtin_prefetch(wpack + (size_t)(((tap + 1) * 4 + wave) * 2 * 32 + lane) * 16, 0, 1);
    __syncthreads();
    wmma_tap(buf, wpack, tap, wave, lane, acc);
  }
  epilogue(acc, dst, nullptr, 0, resid, do_lrelu, wave, half, nn, p0, THW);
}

// ---------------------------------------------------------------------------
// Input conv 3 -> 64 channels (tiny: ~0.3 GFLOP), direct VALU + lrelu.
// ---------------------------------------------------------------------------
__global__ __launch_bounds__(256)
void conv_in_k(const float* __restrict__ x, const float* __restrict__ w,
               float* __restrict__ out) {
  const int HW = 4096, THW = 28672;
  int gid = blockIdx.x * 256 + threadIdx.x;
  if (gid >= 64 * THW) return;
  int o = gid / THW, p = gid - o * THW;
  int t = p >> 12, r = p & 4095, h = r >> 6, ww = r & 63;
  float s = 0.f;
  for (int c = 0; c < 3; ++c)
#pragma unroll
    for (int kt = 0; kt < 3; ++kt) {
      int tt = t + kt - 1;
      if ((unsigned)tt >= 7u) continue;
#pragma unroll
      for (int kh = 0; kh < 3; ++kh) {
        int hh = h + kh - 1;
        if ((unsigned)hh >= 64u) continue;
#pragma unroll
        for (int kw = 0; kw < 3; ++kw) {
          int w2 = ww + kw - 1;
          if ((unsigned)w2 >= 64u) continue;
          s += w[(o * 3 + c) * 27 + kt * 9 + kh * 3 + kw] *
               x[(c * 7 + tt) * HW + (hh << 6) + w2];
        }
      }
    }
  out[gid] = s >= 0.f ? s : 0.1f * s;
}

// maxpool (1,2,2) with row-major (h,w) flat argmax, first-max tie-break
__global__ __launch_bounds__(256)
void maxpool_k(const float* __restrict__ x, float* __restrict__ y,
               int* __restrict__ idx, int T, int H, int W) {
  int H2 = H >> 1, W2 = W >> 1;
  int total = 64 * T * H2 * W2;
  int gid = blockIdx.x * 256 + threadIdx.x;
  if (gid >= total) return;
  int w2 = gid % W2; int q = gid / W2;
  int h2 = q % H2; q /= H2;
  int t = q % T; int c = q / T;
  const float* xp = x + ((c * T + t) * H + 2 * h2) * W + 2 * w2;
  float best = xp[0]; int bj = 0;
  float v = xp[1];     if (v > best) { best = v; bj = 1; }
  v = xp[W];           if (v > best) { best = v; bj = 2; }
  v = xp[W + 1];       if (v > best) { best = v; bj = 3; }
  y[gid] = best; idx[gid] = bj;
}

// maxunpool fused with skip-connection add: out = unpool(pooled, idx) + other
__global__ __launch_bounds__(256)
void unpool_add_k(const float* __restrict__ pooled, const int* __restrict__ idx,
                  const float* __restrict__ other, float* __restrict__ out,
                  int T, int H2, int W2) {
  int H = H2 * 2, W = W2 * 2;
  int total = 64 * T * H * W;
  int gid = blockIdx.x * 256 + threadIdx.x;
  if (gid >= total) return;
  int w = gid % W; int q = gid / W;
  int h = q % H; q /= H;
  int t = q % T; int c = q / T;
  int j = (h & 1) * 2 + (w & 1);
  int pg = ((c * T + t) * H2 + (h >> 1)) * W2 + (w >> 1);
  float v = (idx[pg] == j) ? pooled[pg] : 0.f;
  out[gid] = v + other[gid];
}

__global__ __launch_bounds__(256)
void add_k(const float* __restrict__ a, const float* __restrict__ b,
           float* __restrict__ out, int n) {
  int gid = blockIdx.x * 256 + threadIdx.x;
  if (gid < n) out[gid] = a[gid] + b[gid];
}

// ---------------------------------------------------------------------------
extern "C" void kernel_launch(void* const* d_in, const int* in_sizes, int n_in,
                              void* d_out, int out_size, void* d_ws, size_t ws_size,
                              hipStream_t stream) {
  (void)in_sizes; (void)n_in; (void)out_size; (void)ws_size;
  const float* x     = (const float*)d_in[0];   // [1,3,7,64,64]
  const float* w_in  = (const float*)d_in[1];   // [64,3,3,3,3]
  const float* w_out = (const float*)d_in[2];   // [64,64,3,3,3]
  const float* dcn_w = (const float*)d_in[3];   // [12,64,64,3,3,3]
  const float* off_w = (const float*)d_in[4];   // [12,54,64,3,3,3]
  const float* off_b = (const float*)d_in[5];   // [12,54]

  char* base = (char*)d_ws;
  const size_t FULLB = (size_t)64 * 7 * 64 * 64 * sizeof(float);  // 7,340,032 B
  float* S[7];
  for (int i = 0; i < 7; ++i) S[i] = (float*)(base + (size_t)i * FULLB);
  int* idxA = (int*)(base + 7 * FULLB);                          // 64*7*32*32
  int* idxB = (int*)((char*)idxA + (size_t)64 * 7 * 32 * 32 * sizeof(int));
  __bf16* packs = (__bf16*)((char*)idxB + (size_t)64 * 7 * 16 * 16 * sizeof(int));
  const size_t PK = 27 * 4 * 2 * 32 * 16;                        // elems per pack
  auto dcnpack = [&](int i) { return packs + (size_t)i * PK; };
  auto offpack = [&](int i) { return packs + (size_t)(12 + i) * PK; };
  __bf16* outpack = packs + (size_t)24 * PK;

  // pack all weights into bf16 A-fragment layout (cheap, every launch)
  for (int i = 0; i < 12; ++i) {
    pack_w_k<<<216, 512, 0, stream>>>(dcn_w + (size_t)i * 64 * 64 * 27, dcnpack(i), 64);
    pack_w_k<<<216, 512, 0, stream>>>(off_w + (size_t)i * 54 * 64 * 27, offpack(i), 54);
  }
  pack_w_k<<<216, 512, 0, stream>>>(w_out, outpack, 64);

  float* OFF = S[3];
  float* MID = S[4];

  // inp = lrelu(conv3d(x, w_in)) -> S0
  conv_in_k<<<(64 * 28672 + 255) / 256, 256, 0, stream>>>(x, w_in, S[0]);

  auto resblock = [&](const float* srcA, float* dstA, int i, int T, int lgH, int lgW) {
    int THW = T << (lgH + lgW);
    conv64_wmma_k<<<THW / 16, 128, 0, stream>>>(srcA, offpack(2 * i),
        off_b + 54 * (2 * i), 54, nullptr, 0, OFF, T, lgH, lgW);
    deform_wmma_k<<<THW / 16, 128, 0, stream>>>(srcA, OFF, dcnpack(2 * i),
        nullptr, 1, MID, T, lgH, lgW);
    conv64_wmma_k<<<THW / 16, 128, 0, stream>>>(MID, offpack(2 * i + 1),
        off_b + 54 * (2 * i + 1), 54, nullptr, 0, OFF, T, lgH, lgW);
    deform_wmma_k<<<THW / 16, 128, 0, stream>>>(MID, OFF, dcnpack(2 * i + 1),
        srcA, 0, dstA, T, lgH, lgW);
  };

  const int nf = 64 * 7 * 64 * 64, nh = 64 * 7 * 32 * 32, nq = 64 * 7 * 16 * 16;

  resblock(S[0], S[1], 0, 7, 6, 6);                                     // r1
  maxpool_k<<<(nh + 255) / 256, 256, 0, stream>>>(S[1], S[5], idxA, 7, 64, 64);
  resblock(S[5], S[2], 1, 7, 5, 5);                                     // r2
  maxpool_k<<<(nq + 255) / 256, 256, 0, stream>>>(S[2], S[5], idxB, 7, 32, 32);
  resblock(S[5], S[6], 2, 7, 4, 4);                                     // r3
  resblock(S[6], S[5], 3, 7, 4, 4);                                     // r4
  unpool_add_k<<<(nh + 255) / 256, 256, 0, stream>>>(S[5], idxB, S[2], S[6], 7, 16, 16);
  resblock(S[6], S[5], 4, 7, 5, 5);                                     // r5
  unpool_add_k<<<(nf + 255) / 256, 256, 0, stream>>>(S[5], idxA, S[1], S[6], 7, 32, 32);
  resblock(S[6], S[5], 5, 7, 6, 6);                                     // r6
  add_k<<<(nf + 255) / 256, 256, 0, stream>>>(S[5], S[0], S[6], nf);    // r6 + inp
  conv64_wmma_k<<<28672 / 16, 128, 0, stream>>>(S[6], outpack, nullptr, 0,
      nullptr, 1, (float*)d_out, 7, 6, 6);                              // final
}
